// flatten_conv2d_31327491457653
// MI455X (gfx1250) — compile-verified, matrix operand order checked by reading, not run on hardware
//
#include <hip/hip_runtime.h>

// ---------- CDNA5 WMMA types ----------
typedef __attribute__((ext_vector_type(16))) __bf16 v16bf;
typedef __attribute__((ext_vector_type(2)))  __bf16 v2bf;
typedef __attribute__((ext_vector_type(8)))  float  v8f;

struct alignas(16) Q16 { unsigned int a, b, c, d; };
struct alignas(32) Frag32 { Q16 lo, hi; };

// ---------- async global->LDS copy (CDNA5, tracked by ASYNCcnt) ----------
__device__ __forceinline__ void async_b128_to_lds(void* lds_dst, const void* gsrc) {
    // generic LDS pointer: low 32 bits are the addrspace(3) byte offset
    unsigned loff = (unsigned)(unsigned long long)lds_dst;
    unsigned long long ga = (unsigned long long)gsrc;
    asm volatile("global_load_async_to_lds_b128 %0, %1, off"
                 :: "v"(loff), "v"(ga) : "memory");
}
__device__ __forceinline__ void wait_async_all() {
    asm volatile("s_wait_asynccnt 0x0" ::: "memory");
}

__device__ __forceinline__ v2bf pack_bf16(float a, float b) {
    v2bf p;
    p.x = (__bf16)a;   // lowers to v_cvt_pk_bf16_f32 when paired
    p.y = (__bf16)b;
    return p;
}

// ---------- weight transpose + f32->bf16 convert:  dst[o][c] = src[c][o] ----------
__global__ void wcvt_kernel(const float* __restrict__ src, __bf16* __restrict__ dst,
                            int Cin, int Cout) {
    int i = blockIdx.x * blockDim.x + threadIdx.x;
    if (i >= Cin * Cout) return;
    int o = i / Cin;
    int c = i - o * Cin;
    dst[i] = (__bf16)src[c * Cout + o];
}

// ---------- pointwise conv as batched GEMM via v_wmma_f32_16x16x32_bf16 ----------
// out[b, m, n] = sum_k wT[m, k] * in[b, k, n] + bias[m]
// block tile: M=128, N=64; 8 waves = (msub 0..3) x (nhalf 0..1); each wave owns a
// 32x32 output tile -> 4 WMMAs per K-step with 2x register-level A/B reuse.
__global__ __launch_bounds__(256)
void gemm_pw_wmma(const float* __restrict__ in,          // (B, Cin, N) f32
                  const __bf16* __restrict__ wT,         // (Cout, Cin) bf16, k-contiguous
                  const float* __restrict__ bias,        // (Cout)
                  float* __restrict__ out,               // (B, Cout, N) f32
                  int Cin, int N, int Cout)
{
    __shared__ __bf16 As[128 * 32];   // [mlocal][k]  8 KB
    __shared__ __bf16 Bs[64 * 32];    // [nlocal][k]  4 KB (transposed stage)

    const int tid   = threadIdx.x;
    const int lane  = tid & 31;
    const int wave  = tid >> 5;
    const int msub  = wave & 3;
    const int nhalf = wave >> 2;
    const int lm    = lane & 15;      // fragment row/col
    const int hi    = lane >> 4;      // lane half -> K-group select

    const int b  = blockIdx.z;
    const int m0 = blockIdx.y * 128;
    const int n0 = blockIdx.x * 64;
    const bool inb = (n0 + 64 <= N);  // block-uniform

    v8f acc00 = {}, acc01 = {}, acc10 = {}, acc11 = {};

    // cooperative-load decomposition
    const int arow = tid >> 1;        // 0..127 (A: 2 threads x 32B per 64B row)
    const int aseg = (tid & 1) * 2;   // 0 or 2 (16B segments)
    const int kq   = (tid >> 4) * 2;  // 0,2,..,30 (B: k-pair base)
    const int n4   = (tid & 15) * 4;  // 0,4,..,60 (B: 4 consecutive pixels)

    const float* inB = in + (size_t)b * Cin * N;

    for (int kk = 0; kk < Cin; kk += 32) {
        // --- A tile: async DMA global(bf16) -> LDS, 2 x b128 per thread ---
        {
            const __bf16* src = wT + (size_t)(m0 + arow) * Cin + kk + aseg * 8;
            __bf16* dst = As + arow * 32 + aseg * 8;
            async_b128_to_lds(dst,      src);
            async_b128_to_lds(dst + 8,  src + 8);
        }
        // --- B tile: load 2 k-rows x 4 pixels, packed cvt, store [n][k] dwords ---
        {
            const float* r0 = inB + (size_t)(kk + kq) * N;
            const float* r1 = r0 + N;
            v2bf* d = (v2bf*)(Bs + n4 * 32 + kq);   // (n4, kq); +16 v2bf per +1 n
            if (inb) {
                const float4 f0 = *(const float4*)(r0 + n0 + n4);
                const float4 f1 = *(const float4*)(r1 + n0 + n4);
                d[0]  = pack_bf16(f0.x, f1.x);
                d[16] = pack_bf16(f0.y, f1.y);
                d[32] = pack_bf16(f0.z, f1.z);
                d[48] = pack_bf16(f0.w, f1.w);
            } else {
                #pragma unroll
                for (int i2 = 0; i2 < 4; ++i2) {
                    int n = n0 + n4 + i2;
                    if (n > N - 1) n = N - 1;      // clamp; cols >= N discarded on store
                    d[i2 * 16] = pack_bf16(r0[n], r1[n]);
                }
            }
            if (kk + 32 < Cin)   // prefetch next K-tile of activations
                __builtin_prefetch(r0 + (size_t)32 * N + n0 + n4, 0, 0);
        }
        wait_async_all();
        __syncthreads();

        // --- fragments: lane(lm,hi) -> K runs [8hi,+8) and [16+8hi,+8) ---
        Frag32 fa0, fa1, fb0, fb1;
        const __bf16* Ab0 = As + (msub * 32 + lm) * 32;
        const __bf16* Ab1 = Ab0 + 16 * 32;
        fa0.lo = *(const Q16*)(Ab0 + 8 * hi);
        fa0.hi = *(const Q16*)(Ab0 + 16 + 8 * hi);
        fa1.lo = *(const Q16*)(Ab1 + 8 * hi);
        fa1.hi = *(const Q16*)(Ab1 + 16 + 8 * hi);
        const __bf16* Bb0 = Bs + (nhalf * 32 + lm) * 32;
        const __bf16* Bb1 = Bb0 + 16 * 32;
        fb0.lo = *(const Q16*)(Bb0 + 8 * hi);
        fb0.hi = *(const Q16*)(Bb0 + 16 + 8 * hi);
        fb1.lo = *(const Q16*)(Bb1 + 8 * hi);
        fb1.hi = *(const Q16*)(Bb1 + 16 + 8 * hi);
        v16bf a0 = __builtin_bit_cast(v16bf, fa0);
        v16bf a1 = __builtin_bit_cast(v16bf, fa1);
        v16bf b0 = __builtin_bit_cast(v16bf, fb0);
        v16bf b1 = __builtin_bit_cast(v16bf, fb1);

        acc00 = __builtin_amdgcn_wmma_f32_16x16x32_bf16(false, a0, false, b0,
                                                        (short)0, acc00, false, false);
        acc01 = __builtin_amdgcn_wmma_f32_16x16x32_bf16(false, a0, false, b1,
                                                        (short)0, acc01, false, false);
        acc10 = __builtin_amdgcn_wmma_f32_16x16x32_bf16(false, a1, false, b0,
                                                        (short)0, acc10, false, false);
        acc11 = __builtin_amdgcn_wmma_f32_16x16x32_bf16(false, a1, false, b1,
                                                        (short)0, acc11, false, false);

        __syncthreads();
    }

    // --- epilogue: D layout lanes 0-15 -> M=r, lanes 16-31 -> M=r+8; N=lane%16 ---
    const int mr0 = m0 + msub * 32 + hi * 8;
    const int mr1 = mr0 + 16;
    const int nc0 = n0 + nhalf * 32 + lm;
    const int nc1 = nc0 + 16;
    float* o0 = out + ((size_t)b * Cout + mr0) * N;
    float* o1 = out + ((size_t)b * Cout + mr1) * N;
    #pragma unroll
    for (int r = 0; r < 8; ++r) {
        const float bv0 = bias[mr0 + r];
        const float bv1 = bias[mr1 + r];
        if (nc0 < N) {
            o0[(size_t)r * N + nc0] = acc00[r] + bv0;
            o1[(size_t)r * N + nc0] = acc10[r] + bv1;
        }
        if (nc1 < N) {
            o0[(size_t)r * N + nc1] = acc01[r] + bv0;
            o1[(size_t)r * N + nc1] = acc11[r] + bv1;
        }
    }
}

// ---------- fused separable depthwise (V along H, then H along W), folded bias ----------
// out[h,w] = sum_{i,j} vw[i]*hw[j]*in[h+i-pad, w+j-pad]  (OOB -> 0)  + vb*sum(hw) + hb
__global__ void dw_sep_kernel(const float* __restrict__ in,
                              const float* __restrict__ vw, const float* __restrict__ vb,
                              const float* __restrict__ hw, const float* __restrict__ hb,
                              float* __restrict__ out,
                              int C, int Hin, int Win, int Hout, int Wout, int pad, int total)
{
    int idx = blockIdx.x * blockDim.x + threadIdx.x;
    if (idx >= total) return;
    int w  = idx % Wout;
    int h  = (idx / Wout) % Hout;
    int bc = idx / (Wout * Hout);
    int c  = bc % C;

    float vt[3] = { vw[c * 3 + 0], vw[c * 3 + 1], vw[c * 3 + 2] };
    float ht[3] = { hw[c * 3 + 0], hw[c * 3 + 1], hw[c * 3 + 2] };

    const float* p = in + (size_t)bc * Hin * Win;
    float acc = 0.f;
    #pragma unroll
    for (int i = 0; i < 3; ++i) {
        int r = h + i - pad;
        if (r < 0 || r >= Hin) continue;
        const float* row = p + (size_t)r * Win;
        #pragma unroll
        for (int j = 0; j < 3; ++j) {
            int s = w + j - pad;
            if (s < 0 || s >= Win) continue;
            acc = fmaf(vt[i] * ht[j], row[s], acc);
        }
    }
    out[idx] = acc + vb[c] * (ht[0] + ht[1] + ht[2]) + hb[c];
}

// ---------- launcher ----------
extern "C" void kernel_launch(void* const* d_in, const int* in_sizes, int n_in,
                              void* d_out, int out_size, void* d_ws, size_t ws_size,
                              hipStream_t stream) {
    (void)in_sizes; (void)n_in; (void)out_size; (void)ws_size;

    const float* x   = (const float*)d_in[0];
    const float* L1w = (const float*)d_in[1];
    const float* L1b = (const float*)d_in[2];
    const float* V1w = (const float*)d_in[3];
    const float* V1b = (const float*)d_in[4];
    const float* H1w = (const float*)d_in[5];
    const float* H1b = (const float*)d_in[6];
    const float* L2w = (const float*)d_in[7];
    const float* L2b = (const float*)d_in[8];
    const float* V2w = (const float*)d_in[9];
    const float* V2b = (const float*)d_in[10];
    const float* H2w = (const float*)d_in[11];
    const float* H2b = (const float*)d_in[12];

    const int Bn = 32, CIN = 128, COUT = 256, H = 64, W = 64;
    const int HO = 62, WO = 62;
    const int N1 = H * W;     // 4096
    const int N2 = HO * WO;   // 3844

    char* ws = (char*)d_ws;
    __bf16* wT1 = (__bf16*)(ws);                    // 64 KB
    __bf16* wT2 = (__bf16*)(ws + (64 << 10));       // 128 KB
    float* Y1 = (float*)(ws + (256 << 10));         // 134 MB
    float* Y2 = (float*)(ws + (256 << 10) + (size_t)Bn * COUT * N1 * sizeof(float));
    float* Y3 = Y1;                                 // reuse Y1 region (no aliasing in order)

    // 1) weight transpose+convert to bf16
    wcvt_kernel<<<(CIN * COUT + 255) / 256, 256, 0, stream>>>(L1w, wT1, CIN, COUT);
    wcvt_kernel<<<(COUT * COUT + 255) / 256, 256, 0, stream>>>(L2w, wT2, COUT, COUT);

    // 2) L1 pointwise GEMM (+bias) -> Y1 (B,256,64,64)
    gemm_pw_wmma<<<dim3(N1 / 64, COUT / 128, Bn), 256, 0, stream>>>(
        x, wT1, L1b, Y1, CIN, N1, COUT);

    // 3) V1 then H1 fused -> Y2 (B,256,62,62)
    const int tot = Bn * COUT * HO * WO;
    dw_sep_kernel<<<(tot + 255) / 256, 256, 0, stream>>>(
        Y1, V1w, V1b, H1w, H1b, Y2, COUT, H, W, HO, WO, 0, tot);

    // 4) L2 pointwise GEMM (+bias) -> Y3 (B,256,62,62)
    gemm_pw_wmma<<<dim3((N2 + 63) / 64, COUT / 128, Bn), 256, 0, stream>>>(
        Y2, wT2, L2b, Y3, COUT, N2, COUT);

    // 5) zero-pad + V2 then H2 fused -> d_out (B,256,62,62)
    dw_sep_kernel<<<(tot + 255) / 256, 256, 0, stream>>>(
        Y3, V2w, V2b, H2w, H2b, (float*)d_out, COUT, HO, WO, HO, WO, 1, tot);
}
